// Encoder_28028956574172
// MI455X (gfx1250) — matrix operand order repaired and checked by reading, not compiled, not run on hardware
//
#include <hip/hip_runtime.h>

// ---------------------------------------------------------------------------
// Problem constants (from reference)
// ---------------------------------------------------------------------------
#define B_SZ   512
#define S_LEN  1024
#define DIN    64
#define DC     16
#define DXC    80          // DIN + DC
#define DXP    96          // padded to multiple of 32 for WMMA K
#define H      256
#define G3H    768         // 3 * H
#define REP    128

// bf16 workspace layout (element offsets)
#define OFF_W1    0                         // W_ih1 padded  [768][96]
#define OFF_WHH1  (OFF_W1   + G3H * DXP)    // W_hh1         [768][256]
#define OFF_WIH2  (OFF_WHH1 + G3H * H)      // W_ih2         [768][256]
#define OFF_WHH2  (OFF_WIH2 + G3H * H)      // W_hh2         [768][256]
#define OFF_WLIN  (OFF_WHH2 + G3H * H)      // W_lin         [128][256]
#define WS_ELEMS  (OFF_WLIN + REP * H)      // 696,320 bf16 ~= 1.36 MB

typedef __attribute__((ext_vector_type(16))) __bf16 v16bf;
typedef __attribute__((ext_vector_type(8)))  __bf16 v8bf;
typedef __attribute__((ext_vector_type(8)))  float  v8f;

// ---------------------------------------------------------------------------
// Prep kernel: convert fp32 weights -> bf16 into workspace (W_ih1 zero-padded
// from K=80 to K=96 so layer-1 GEMM uses 3 clean K=32 tiles).
// ---------------------------------------------------------------------------
__global__ void gru_cvt_weights(const float* __restrict__ Wih1,
                                const float* __restrict__ Whh1,
                                const float* __restrict__ Wih2,
                                const float* __restrict__ Whh2,
                                const float* __restrict__ Wlin,
                                __bf16* __restrict__ ws) {
    int i = blockIdx.x * blockDim.x + threadIdx.x;
    if (i >= WS_ELEMS) return;
    if (i < OFF_WHH1) {                       // padded W_ih1
        int r = i / DXP, c = i % DXP;
        ws[i] = (__bf16)(c < DXC ? Wih1[r * DXC + c] : 0.0f);
    } else if (i < OFF_WIH2) {
        ws[i] = (__bf16)Whh1[i - OFF_WHH1];
    } else if (i < OFF_WHH2) {
        ws[i] = (__bf16)Wih2[i - OFF_WIH2];
    } else if (i < OFF_WLIN) {
        ws[i] = (__bf16)Whh2[i - OFF_WHH2];
    } else {
        ws[i] = (__bf16)Wlin[i - OFF_WLIN];
    }
}

// ---------------------------------------------------------------------------
// WMMA fragment helpers (ISA 7.12.2 layouts, wave32)
// ---------------------------------------------------------------------------

// A-fragment: 16x32 bf16, lane l holds row m=l&15; lanes 0-15 K={kt..kt+7,
// kt+16..kt+23}, lanes 16-31 K={kt+8..kt+15, kt+24..kt+31}.
__device__ __forceinline__ v16bf load_a_frag(const __bf16* base, int ldk,
                                             int kt, int lane) {
    int m  = lane & 15;
    int kb = (lane >> 4) << 3;          // 0 or 8
    const __bf16* row = base + m * ldk + kt + kb;
    v8bf lo = *reinterpret_cast<const v8bf*>(row);
    v8bf hi = *reinterpret_cast<const v8bf*>(row + 16);
    return __builtin_shufflevector(lo, hi, 0, 1, 2, 3, 4, 5, 6, 7,
                                           8, 9, 10, 11, 12, 13, 14, 15);
}

// B-fragment: 32x16 bf16 (K x N). Source W is row-major [N_total][ldk]
// (out = A @ W^T), so W row (n0+n) IS column n: lanes 0-15 K={kt..kt+7,
// kt+16..kt+23}, lanes 16-31 K={kt+8..kt+15, kt+24..kt+31}.
__device__ __forceinline__ v16bf load_b_frag(const __bf16* W, int ldk,
                                             int n0, int kt, int lane) {
    int n    = lane & 15;
    int koff = (lane >> 4) << 3;        // 0 or 8
    const __bf16* col = W + (size_t)(n0 + n) * ldk + kt + koff;
    v8bf lo = *reinterpret_cast<const v8bf*>(col);
    v8bf hi = *reinterpret_cast<const v8bf*>(col + 16);
    return __builtin_shufflevector(lo, hi, 0, 1, 2, 3, 4, 5, 6, 7,
                                           8, 9, 10, 11, 12, 13, 14, 15);
}

// C/D fragment scatter: VGPR i -> row (i + 8*(lane>=16)), col lane&15.
__device__ __forceinline__ void store_c_frag(float* dst, int ldn, int n0,
                                             v8f c, int lane) {
    int col    = lane & 15;
    int rowoff = (lane >> 4) << 3;
#pragma unroll
    for (int i = 0; i < 8; ++i)
        dst[(rowoff + i) * ldn + n0 + col] = c[i];
}

__device__ __forceinline__ float sigm(float x) {
    return 1.0f / (1.0f + __expf(-x));
}
__device__ __forceinline__ float fast_tanh(float x) {
    // tanh(x) = 1 - 2/(exp(2x)+1)   (single v_exp_f32 TRANS op)
    return 1.0f - 2.0f / (__expf(2.0f * x) + 1.0f);
}

// ---------------------------------------------------------------------------
// Persistent GRU kernel: one block owns 16 batch rows for all 1024 steps.
// 512 threads = 16 waves; each wave owns 3 of the 48 N-tiles (768 gates).
// LDS = 147 KB -> two blocks co-resident per WGP (294 KB <= 320 KB).
// ---------------------------------------------------------------------------
__global__ __launch_bounds__(512)
void gru_persistent(const float* __restrict__ input,   // [B,S,64]
                    const float* __restrict__ cond,    // [B,S,16]
                    const float* __restrict__ b_ih1,   // [768]
                    const float* __restrict__ b_hh1,   // [768]
                    const float* __restrict__ b_ih2,   // [768]
                    const float* __restrict__ b_hh2,   // [768]
                    const float* __restrict__ b_lin,   // [128]
                    const __bf16* __restrict__ ws,     // converted weights
                    float* __restrict__ out) {         // [B,128]
    __shared__ __bf16 sX [16][DXP];    //  3 KB   x_t tile (bf16, zero-padded)
    __shared__ __bf16 sH1[16][H];      //  8 KB   h1 (bf16 WMMA operand)
    __shared__ __bf16 sH2[16][H];      //  8 KB   h2 (bf16 WMMA operand)
    __shared__ float  fH1[16][H];      // 16 KB   h1 (fp32 master state)
    __shared__ float  fH2[16][H];      // 16 KB   h2 (fp32 master state)
    __shared__ float  sGI[16][G3H];    // 48 KB   x/h1-side gate pre-acts
    __shared__ float  sGH[16][G3H];    // 48 KB   h-side gate pre-acts

    const int tid  = threadIdx.x;
    const int lane = tid & 31;
    const int wave = tid >> 5;                 // 0..15
    const int m0   = blockIdx.x * 16;          // batch row base
    const int nbase = wave * 3 * 16;           // this wave's first N column

    // h0 = 0
    for (int i = tid; i < 16 * H; i += 512) {
        int m = i >> 8, j = i & (H - 1);
        fH1[m][j] = 0.0f; fH2[m][j] = 0.0f;
        sH1[m][j] = (__bf16)0.0f; sH2[m][j] = (__bf16)0.0f;
    }
    __syncthreads();

    for (int t = 0; t < S_LEN; ++t) {
        // Opaque *offset* (not pointer): defeats LICM hoist-and-spill of the
        // weight streams across timesteps while keeping the GEP rooted at the
        // global kernel-arg pointer, so loads lower to global_load_b128
        // (LOADcnt-only VMEM path), not flat_load (LOADcnt+DScnt).
        size_t woff = 0;
        asm volatile("" : "+s"(woff));
        const __bf16* W1   = ws + OFF_W1   + woff;
        const __bf16* Whh1 = ws + OFF_WHH1 + woff;
        const __bf16* Wih2 = ws + OFF_WIH2 + woff;
        const __bf16* Whh2 = ws + OFF_WHH2 + woff;

        // ---- stage x_t tile into LDS as bf16 (padded K=96) ----
        for (int i = tid; i < 16 * DXP; i += 512) {
            int m = i / DXP, c = i % DXP;
            float v = 0.0f;
            if (c < DIN)      v = input[((size_t)(m0 + m) * S_LEN + t) * DIN + c];
            else if (c < DXC) v = cond [((size_t)(m0 + m) * S_LEN + t) * DC  + (c - DIN)];
            sX[m][c] = (__bf16)v;
        }
        // Prefetch next step's x/cond rows (global_prefetch_b8).
        if (t + 1 < S_LEN) {
            if (tid < 32)
                __builtin_prefetch(&input[((size_t)(m0 + (tid >> 1)) * S_LEN + (t + 1)) * DIN
                                          + (tid & 1) * 32], 0, 1);
            else if (tid < 48)
                __builtin_prefetch(&cond[((size_t)(m0 + (tid - 32)) * S_LEN + (t + 1)) * DC],
                                   0, 1);
        }
        __syncthreads();

        // ---- layer-1 GEMMs: gi1 = x @ W_ih1^T ; gh1 = h1 @ W_hh1^T ----
#pragma clang loop unroll(disable)
        for (int nt = 0; nt < 3; ++nt) {
            const int n0 = nbase + nt * 16;
            v8f gi = {};
#pragma unroll
            for (int k = 0; k < DXP / 32; ++k) {
                v16bf a = load_a_frag(&sX[0][0], DXP, k * 32, lane);
                v16bf b = load_b_frag(W1, DXP, n0, k * 32, lane);
                gi = __builtin_amdgcn_wmma_f32_16x16x32_bf16(
                         false, a, false, b, (short)0, gi, false, false);
            }
            store_c_frag(&sGI[0][0], G3H, n0, gi, lane);
            v8f gh = {};
#pragma unroll
            for (int k = 0; k < H / 32; ++k) {
                v16bf a = load_a_frag(&sH1[0][0], H, k * 32, lane);
                v16bf b = load_b_frag(Whh1, H, n0, k * 32, lane);
                gh = __builtin_amdgcn_wmma_f32_16x16x32_bf16(
                         false, a, false, b, (short)0, gh, false, false);
            }
            store_c_frag(&sGH[0][0], G3H, n0, gh, lane);
        }
        __syncthreads();

        // ---- h1 elementwise update (fp32 master state) ----
        for (int i = tid; i < 16 * H; i += 512) {
            int m = i >> 8, j = i & (H - 1);
            float r  = sigm(sGI[m][j]       + b_ih1[j]       + sGH[m][j]       + b_hh1[j]);
            float z  = sigm(sGI[m][j + H]   + b_ih1[j + H]   + sGH[m][j + H]   + b_hh1[j + H]);
            float nn = fast_tanh(sGI[m][j + 2*H] + b_ih1[j + 2*H]
                                 + r * (sGH[m][j + 2*H] + b_hh1[j + 2*H]));
            float hn = (1.0f - z) * nn + z * fH1[m][j];
            fH1[m][j] = hn;
            sH1[m][j] = (__bf16)hn;
        }
        __syncthreads();

        // ---- layer-2 GEMMs: gi2 = h1 @ W_ih2^T ; gh2 = h2(t-1) @ W_hh2^T ----
#pragma clang loop unroll(disable)
        for (int nt = 0; nt < 3; ++nt) {
            const int n0 = nbase + nt * 16;
            v8f gi = {};
#pragma unroll
            for (int k = 0; k < H / 32; ++k) {
                v16bf a = load_a_frag(&sH1[0][0], H, k * 32, lane);
                v16bf b = load_b_frag(Wih2, H, n0, k * 32, lane);
                gi = __builtin_amdgcn_wmma_f32_16x16x32_bf16(
                         false, a, false, b, (short)0, gi, false, false);
            }
            store_c_frag(&sGI[0][0], G3H, n0, gi, lane);
            v8f gh = {};
#pragma unroll
            for (int k = 0; k < H / 32; ++k) {
                v16bf a = load_a_frag(&sH2[0][0], H, k * 32, lane);
                v16bf b = load_b_frag(Whh2, H, n0, k * 32, lane);
                gh = __builtin_amdgcn_wmma_f32_16x16x32_bf16(
                         false, a, false, b, (short)0, gh, false, false);
            }
            store_c_frag(&sGH[0][0], G3H, n0, gh, lane);
        }
        __syncthreads();

        // ---- h2 elementwise update ----
        for (int i = tid; i < 16 * H; i += 512) {
            int m = i >> 8, j = i & (H - 1);
            float r  = sigm(sGI[m][j]       + b_ih2[j]       + sGH[m][j]       + b_hh2[j]);
            float z  = sigm(sGI[m][j + H]   + b_ih2[j + H]   + sGH[m][j + H]   + b_hh2[j + H]);
            float nn = fast_tanh(sGI[m][j + 2*H] + b_ih2[j + 2*H]
                                 + r * (sGH[m][j + 2*H] + b_hh2[j + 2*H]));
            float hn = (1.0f - z) * nn + z * fH2[m][j];
            fH2[m][j] = hn;
            sH2[m][j] = (__bf16)hn;
        }
        __syncthreads();
    }

    // ---- final linear: out = h2 @ W_lin^T + b_lin  (REP=128 -> 8 N-tiles) ----
    if (wave < 8) {
        const __bf16* Wlin = ws + OFF_WLIN;
        const int n0 = wave * 16;
        v8f acc = {};
#pragma unroll
        for (int k = 0; k < H / 32; ++k) {
            v16bf a = load_a_frag(&sH2[0][0], H, k * 32, lane);
            v16bf b = load_b_frag(Wlin, H, n0, k * 32, lane);
            acc = __builtin_amdgcn_wmma_f32_16x16x32_bf16(
                      false, a, false, b, (short)0, acc, false, false);
        }
        int col    = lane & 15;
        int rowoff = (lane >> 4) << 3;
        float bias = b_lin[n0 + col];
#pragma unroll
        for (int i = 0; i < 8; ++i)
            out[(size_t)(m0 + rowoff + i) * REP + n0 + col] = acc[i] + bias;
    }
}

// ---------------------------------------------------------------------------
// Launch
// ---------------------------------------------------------------------------
extern "C" void kernel_launch(void* const* d_in, const int* in_sizes, int n_in,
                              void* d_out, int out_size, void* d_ws, size_t ws_size,
                              hipStream_t stream) {
    const float* input = (const float*)d_in[0];
    const float* cond  = (const float*)d_in[1];
    const float* Wih1  = (const float*)d_in[2];
    const float* Whh1  = (const float*)d_in[3];
    const float* bih1  = (const float*)d_in[4];
    const float* bhh1  = (const float*)d_in[5];
    const float* Wih2  = (const float*)d_in[6];
    const float* Whh2  = (const float*)d_in[7];
    const float* bih2  = (const float*)d_in[8];
    const float* bhh2  = (const float*)d_in[9];
    const float* Wlin  = (const float*)d_in[10];
    const float* blin  = (const float*)d_in[11];
    __bf16* wsb = (__bf16*)d_ws;
    float*  out = (float*)d_out;

    gru_cvt_weights<<<(WS_ELEMS + 255) / 256, 256, 0, stream>>>(
        Wih1, Whh1, Wih2, Whh2, Wlin, wsb);

    gru_persistent<<<B_SZ / 16, 512, 0, stream>>>(
        input, cond, bih1, bhh1, bih2, bhh2, blin, wsb, out);
}
